// MultiHeadAttention_25606595019014
// MI455X (gfx1250) — matrix operand order repaired and checked by reading, not compiled, not run on hardware
//
#include <hip/hip_runtime.h>

// ---------------------------------------------------------------------------
// MI455X / gfx1250 (CDNA5, wave32) implementation of the Graphormer-style
// multi-head attention reference. All dense GEMMs go through
// v_wmma_f32_16x16x32_bf16; softmax/bias-gather/bin-scatter are fused in LDS.
// This revision adds global_prefetch_b8 hints (probe-confirmed builtin) for
// the streaming K/V/weight operands, two k-steps ahead of consumption.
// ---------------------------------------------------------------------------

typedef unsigned short bf16_t;
typedef __attribute__((ext_vector_type(16))) __bf16 v16bf;
typedef __attribute__((ext_vector_type(8)))  float  v8f;

#define H_     8
#define D_     64
#define HID_   512
#define B_     8
#define S_     512
#define THOP_  32
#define TEDGE_ 16
#define SCALE_ 0.125f   // D^-0.5

union Frag { v16bf v; unsigned int u[8]; };

__device__ __forceinline__ unsigned short f2bf(float f) {
  unsigned int u = __float_as_uint(f);
  u += 0x7FFFu + ((u >> 16) & 1u);          // round-to-nearest-even
  return (unsigned short)(u >> 16);
}
__device__ __forceinline__ float bf2f(unsigned short h) {
  return __uint_as_float(((unsigned int)h) << 16);
}

// A-matrix fragment (16x32 bf16, M x K), row-major source, row stride rs elems.
// ISA layout: lanes 0-15 VGPR0-3 -> K=0..7, VGPR4-7 -> K=16..23; lanes 16-31 +8.
__device__ __forceinline__ Frag load_a(const bf16_t* base, long row0, int rs,
                                       int k0, int lane) {
  Frag f;
  const bf16_t* rp = base + (row0 + (lane & 15)) * (long)rs;
  int hk = (lane >> 4) << 3;
#pragma unroll
  for (int v = 0; v < 8; ++v) {
    int k = k0 + ((v & 4) << 2) + hk + ((v & 3) << 1);
    f.u[v] = *reinterpret_cast<const unsigned int*>(rp + k);
  }
  return f;
}

// B-matrix fragment (32x16 bf16, K x N) loaded from B^T stored [n][k] row-major.
// ISA layout: lanes 0-15 VGPR0-3 -> K=0..7, VGPR4-7 -> K=16..23; lanes 16-31 +8.
__device__ __forceinline__ Frag load_b(const bf16_t* base, long n0, int rs,
                                       int k0, int lane) {
  Frag f;
  const bf16_t* rp = base + (n0 + (lane & 15)) * (long)rs;
  int hk = (lane >> 4) << 3;
#pragma unroll
  for (int v = 0; v < 8; ++v) {
    int k = k0 + ((v & 4) << 2) + hk + ((v & 3) << 1);
    f.u[v] = *reinterpret_cast<const unsigned int*>(rp + k);
  }
  return f;
}

// A-fragment from LDS where bf16 pairs are packed in uints, row stride 256 uints.
__device__ __forceinline__ Frag load_a_lds(const unsigned int* Pu, int row0,
                                           int k0, int lane) {
  Frag f;
  const unsigned int* rp = Pu + (row0 + (lane & 15)) * 256;
  int hk = (lane >> 4) << 3;
#pragma unroll
  for (int v = 0; v < 8; ++v) {
    int k = k0 + ((v & 4) << 2) + hk + ((v & 3) << 1);
    f.u[v] = rp[k >> 1];
  }
  return f;
}

__device__ __forceinline__ v8f wmma_bf16(const Frag& a, const Frag& b, v8f c) {
  return __builtin_amdgcn_wmma_f32_16x16x32_bf16(false, a.v, false, b.v,
                                                 (short)0, c, false, false);
}

// ---------------------------------------------------------------------------
// Elementwise f32 -> bf16 convert
__global__ __launch_bounds__(256) void k_cvt(const float* __restrict__ x,
                                             bf16_t* __restrict__ y, int n) {
  int i = blockIdx.x * 256 + threadIdx.x;
  if (i < n) y[i] = f2bf(x[i]);
}

// Transpose [K][N] f32 -> [N][K] bf16 (weights)
__global__ __launch_bounds__(256) void k_transpose(const float* __restrict__ W,
                                                   bf16_t* __restrict__ Wt,
                                                   int K, int N) {
  int i = blockIdx.x * 256 + threadIdx.x;
  if (i < K * N) {
    int k = i / N, n = i - k * N;
    Wt[n * K + k] = f2bf(W[i]);
  }
}

// ---------------------------------------------------------------------------
// 64x64-tile WMMA GEMM:  C[M,N] = A_bf16[M,K] @ Bt_bf16([N][K])^T + bias[N]
// mode 0: store bf16, layout [B,H,S,D]    (row=b*S+s, col=h*D+d)
// mode 2: store bf16, layout [B,H,D,S]    (V transposed for P@V)
// mode 1: store f32 row-major [M][N]      (final projection into d_out)
__global__ __launch_bounds__(128) void k_gemm(const bf16_t* __restrict__ A,
                                              const bf16_t* __restrict__ Bt,
                                              const float* __restrict__ bias,
                                              void* __restrict__ out,
                                              int M, int N, int K, int mode) {
  int tid = threadIdx.x, lane = tid & 31, wave = tid >> 5;
  long wm = (long)blockIdx.x * 64 + (wave >> 1) * 32;
  long wn = (long)blockIdx.y * 64 + (wave & 1) * 32;
  v8f acc[2][2] = {};
  for (int k0 = 0; k0 < K; k0 += 32) {
    // prefetch operand rows two k-steps ahead (L2-resident -> WGP$)
    int kp = k0 + 64;
    if (kp < K) {
      __builtin_prefetch(A + (wm + (lane & 15)) * (long)K + kp, 0, 3);
      __builtin_prefetch(A + (wm + 16 + (lane & 15)) * (long)K + kp, 0, 3);
      __builtin_prefetch(Bt + (wn + (lane & 15)) * (long)K + kp, 0, 3);
      __builtin_prefetch(Bt + (wn + 16 + (lane & 15)) * (long)K + kp, 0, 3);
    }
    Frag a0 = load_a(A, wm, K, k0, lane);
    Frag a1 = load_a(A, wm + 16, K, k0, lane);
    Frag b0 = load_b(Bt, wn, K, k0, lane);
    Frag b1 = load_b(Bt, wn + 16, K, k0, lane);
    acc[0][0] = wmma_bf16(a0, b0, acc[0][0]);
    acc[0][1] = wmma_bf16(a0, b1, acc[0][1]);
    acc[1][0] = wmma_bf16(a1, b0, acc[1][0]);
    acc[1][1] = wmma_bf16(a1, b1, acc[1][1]);
  }
  int n16 = lane & 15, mh = (lane >> 4) << 3;
#pragma unroll
  for (int mi = 0; mi < 2; ++mi) {
#pragma unroll
    for (int ni = 0; ni < 2; ++ni) {
      long ncol = wn + ni * 16 + n16;
      float bv = bias ? bias[ncol] : 0.f;
#pragma unroll
      for (int v = 0; v < 8; ++v) {
        long mrow = wm + mi * 16 + mh + v;
        float val = acc[mi][ni][v] + bv;
        if (mode == 1) {
          ((float*)out)[mrow * N + ncol] = val;
        } else {
          long b = mrow >> 9, s = mrow & 511;   // S_ = 512
          long h = ncol >> 6, d = ncol & 63;    // D_ = 64
          long idx = (mode == 0) ? (((b * H_ + h) * S_ + s) * D_ + d)
                                 : (((b * H_ + h) * D_ + d) * S_ + s);
          ((bf16_t*)out)[idx] = f2bf(val);
        }
      }
    }
  }
}

// ---------------------------------------------------------------------------
// Per-query-row bias score tables:
//   tbl[b,h,s, t<32     ] = qh[b,h,s]·q_hop[h,t] + kh[b,h,s]·k_hop[h,t]
//   tbl[b,h,s, 32+t<48  ] = qh[b,h,s]·q_edge[h,t] + kh[b,h,s]·k_edge[h,t]
__global__ __launch_bounds__(64) void k_scores(const bf16_t* __restrict__ qh,
                                               const bf16_t* __restrict__ kh,
                                               const float* __restrict__ qhopE,
                                               const float* __restrict__ qedgeE,
                                               const float* __restrict__ khopE,
                                               const float* __restrict__ kedgeE,
                                               float* __restrict__ tbl) {
  __shared__ float qr[64], kr[64];
  long bhs = blockIdx.x;                     // (b*H + h)*S + s
  int h = (int)((bhs >> 9) & 7);
  int t = threadIdx.x;
  qr[t] = bf2f(qh[bhs * 64 + t]);
  kr[t] = bf2f(kh[bhs * 64 + t]);
  __syncthreads();
  if (t < 48) {
    int hop = t < 32;
    int ti = hop ? t : t - 32;
    const float* eq = (hop ? qhopE : qedgeE) + ti * HID_ + h * D_;
    const float* ek = (hop ? khopE : kedgeE) + ti * HID_ + h * D_;
    float s = 0.f;
#pragma unroll 8
    for (int d = 0; d < 64; ++d) s += qr[d] * eq[d] + kr[d] * ek[d];
    tbl[bhs * 48 + t] = s;
  }
}

// ---------------------------------------------------------------------------
// Fused attention: per (b, h, 64-row query tile):
//   QK^T (WMMA) + bias gather + mask -> softmax in LDS -> bins (ds_add_f32)
//   -> in-place bf16 convert -> P@V (WMMA) + bins@value_emb -> xcat bf16
__global__ __launch_bounds__(256) void k_attn(
    const bf16_t* __restrict__ qh, const bf16_t* __restrict__ kh,
    const bf16_t* __restrict__ vT, const float* __restrict__ tbl,
    const int* __restrict__ dist, const int* __restrict__ edge,
    const unsigned char* __restrict__ mask,
    const unsigned char* __restrict__ focal,
    const float* __restrict__ vhopE, const float* __restrict__ vedgeE,
    bf16_t* __restrict__ xcat) {
  extern __shared__ char smem[];
  float* logits = (float*)smem;                 // 64 x 512 f32 (reused as bf16)
  unsigned int* Pu = (unsigned int*)smem;       // same storage, packed bf16
  float* tb = (float*)(smem + 64 * 512 * 4);    // 64 x 48 (scores, then bins)

  int blk = blockIdx.x;                         // B*H*(S/64) = 512
  int qt = blk & 7, h = (blk >> 3) & 7, b = blk >> 6;
  int q0 = qt * 64;
  int tid = threadIdx.x, lane = tid & 31, wave = tid >> 5;
  long bh = b * H_ + h;
  const bf16_t* Q  = qh + bh * (long)(S_ * D_);
  const bf16_t* Kb = kh + bh * (long)(S_ * D_);
  const bf16_t* Vt = vT + bh * (long)(D_ * S_);
  const int* distB = dist + (long)b * S_ * S_;
  const int* edgeB = edge + (long)b * S_ * S_;

  for (int i = tid; i < 64 * 48; i += 256)
    tb[i] = tbl[(bh * S_ + q0) * 48 + i];
  __syncthreads();

  // ---- Phase 1: logits = (QK^T + hop + edge) * scale, masked --------------
  {
    int mt = wave & 3;         // m-tile 0..3
    int jh = wave >> 2;        // j-tile half
    Frag qa0 = load_a(Q, q0 + mt * 16, D_, 0, lane);
    Frag qa1 = load_a(Q, q0 + mt * 16, D_, 32, lane);
    int n16 = lane & 15, mh = (lane >> 4) << 3;
    for (int jt = jh * 16; jt < jh * 16 + 16; ++jt) {
      int j0 = jt * 16;
      // prefetch next K tile rows while this tile computes
      if (jt + 1 < jh * 16 + 16)
        __builtin_prefetch(Kb + (long)(j0 + 16 + (lane & 15)) * D_, 0, 3);
      Frag kb0 = load_b(Kb, j0, D_, 0, lane);
      Frag kb1 = load_b(Kb, j0, D_, 32, lane);
      v8f c = {};
      c = wmma_bf16(qa0, kb0, c);
      c = wmma_bf16(qa1, kb1, c);
      int j = j0 + n16;
      bool km = mask[b * S_ + j] != 0;
#pragma unroll
      for (int v = 0; v < 8; ++v) {
        int m = mt * 16 + mh + v;
        int qrow = q0 + m;
        int di = distB[qrow * S_ + j];
        int ei = edgeB[qrow * S_ + j];
        float val = (c[v] + tb[m * 48 + di] + tb[m * 48 + 32 + ei]) * SCALE_;
        bool ok = (h < 4) ? km : (focal[((long)b * S_ + qrow) * S_ + j] != 0);
        logits[m * 512 + j] = ok ? val : -1e30f;
      }
    }
  }
  __syncthreads();

  // ---- Phase 2: row softmax (4 threads per row) ---------------------------
  int row = tid >> 2, l4 = tid & 3;
  float* lr = logits + row * 512;
  float mx = -1e30f;
  for (int j = l4; j < 512; j += 4) mx = fmaxf(mx, lr[j]);
  tb[row * 48 + l4] = mx;
  __syncthreads();
  mx = fmaxf(fmaxf(tb[row * 48 + 0], tb[row * 48 + 1]),
             fmaxf(tb[row * 48 + 2], tb[row * 48 + 3]));
  __syncthreads();
  float sum = 0.f;
  for (int j = l4; j < 512; j += 4) {
    float e = __expf(lr[j] - mx);
    lr[j] = e;
    sum += e;
  }
  tb[row * 48 + l4] = sum;
  __syncthreads();
  sum = tb[row * 48 + 0] + tb[row * 48 + 1] + tb[row * 48 + 2] + tb[row * 48 + 3];
  float inv = 1.f / sum;
  for (int j = l4; j < 512; j += 4) lr[j] *= inv;
  __syncthreads();

  // ---- Phase 3: scatter attn weights into hop/edge bins (ds_add_f32) -----
  for (int i = tid; i < 64 * 48; i += 256) tb[i] = 0.f;
  __syncthreads();
  {
    int qrow = q0 + row;
    const int* dr = distB + qrow * S_;
    const int* er = edgeB + qrow * S_;
    for (int j = l4; j < 512; j += 4) {
      float p = lr[j];
      atomicAdd(&tb[row * 48 + dr[j]], p);
      atomicAdd(&tb[row * 48 + 32 + er[j]], p);
    }
  }
  __syncthreads();

  // ---- Phase 4: in-place f32 -> packed bf16 convert (chunked, hazard-safe)
  for (int c0 = 0; c0 < 64 * 512; c0 += 2048) {
    float vals[8];
#pragma unroll
    for (int i = 0; i < 8; ++i) vals[i] = logits[c0 + tid * 8 + i];
    __syncthreads();
#pragma unroll
    for (int i = 0; i < 4; ++i) {
      unsigned int pk = (unsigned int)f2bf(vals[2 * i]) |
                        ((unsigned int)f2bf(vals[2 * i + 1]) << 16);
      Pu[(c0 >> 1) + tid * 4 + i] = pk;
    }
    __syncthreads();
  }

  // ---- Phase 5: context = P@V (WMMA) + bins @ value embeddings -----------
  {
    int mt = wave >> 1;           // m-tile 0..3
    int np = (wave & 1) * 2;      // n-tile pair
    v8f acc0 = {}, acc1 = {};
    for (int k0 = 0; k0 < 512; k0 += 32) {
      // prefetch next V tile rows (stride S_) while this step computes
      int kp = k0 + 64;
      if (kp < 512) {
        __builtin_prefetch(Vt + (long)(np * 16 + (lane & 15)) * S_ + kp, 0, 3);
        __builtin_prefetch(Vt + (long)((np + 1) * 16 + (lane & 15)) * S_ + kp,
                           0, 3);
      }
      Frag pa  = load_a_lds(Pu, mt * 16, k0, lane);
      Frag vb0 = load_b(Vt, (long)(np + 0) * 16, S_, k0, lane);
      Frag vb1 = load_b(Vt, (long)(np + 1) * 16, S_, k0, lane);
      acc0 = wmma_bf16(pa, vb0, acc0);
      acc1 = wmma_bf16(pa, vb1, acc1);
    }
    int n16 = lane & 15, mh = (lane >> 4) << 3;
#pragma unroll
    for (int t2 = 0; t2 < 2; ++t2) {
      v8f a = t2 ? acc1 : acc0;
      int d = (np + t2) * 16 + n16;
      float add[8] = {0, 0, 0, 0, 0, 0, 0, 0};
      const float* eh = vhopE + h * D_ + d;
      const float* ee = vedgeE + h * D_ + d;
      for (int t = 0; t < THOP_; ++t) {
        float ev = eh[t * HID_];
#pragma unroll
        for (int v = 0; v < 8; ++v)
          add[v] += tb[(mt * 16 + mh + v) * 48 + t] * ev;
      }
      for (int t = 0; t < TEDGE_; ++t) {
        float ev = ee[t * HID_];
#pragma unroll
        for (int v = 0; v < 8; ++v)
          add[v] += tb[(mt * 16 + mh + v) * 48 + 32 + t] * ev;
      }
#pragma unroll
      for (int v = 0; v < 8; ++v) {
        int m = mt * 16 + mh + v;
        long s = q0 + m;
        xcat[((long)b * S_ + s) * HID_ + h * D_ + d] = f2bf(a[v] + add[v]);
      }
    }
  }
}

// ---------------------------------------------------------------------------
extern "C" void kernel_launch(void* const* d_in, const int* in_sizes, int n_in,
                              void* d_out, int out_size, void* d_ws,
                              size_t ws_size, hipStream_t stream) {
  (void)in_sizes; (void)n_in; (void)out_size; (void)ws_size;
  const float* q      = (const float*)d_in[0];
  const float* k      = (const float*)d_in[1];
  const float* v      = (const float*)d_in[2];
  const float* qhopE  = (const float*)d_in[3];
  const float* qedgeE = (const float*)d_in[4];
  const float* khopE  = (const float*)d_in[5];
  const float* kedgeE = (const float*)d_in[6];
  const float* vhopE  = (const float*)d_in[7];
  const float* vedgeE = (const float*)d_in[8];
  const int* dist     = (const int*)d_in[9];
  const int* edge     = (const int*)d_in[10];
  const unsigned char* mask  = (const unsigned char*)d_in[11];
  const unsigned char* focal = (const unsigned char*)d_in[12];
  const float* Wq = (const float*)d_in[13];
  const float* bq = (const float*)d_in[14];
  const float* Wk = (const float*)d_in[15];
  const float* bk = (const float*)d_in[16];
  const float* Wv = (const float*)d_in[17];
  const float* bv = (const float*)d_in[18];
  const float* Wo = (const float*)d_in[19];
  const float* bo = (const float*)d_in[20];

  const long NE = (long)B_ * S_ * HID_;     // 2,097,152 activations
  const long WE = (long)HID_ * HID_;        // 262,144 weight elements
  char* w = (char*)d_ws;
  bf16_t* qbf = (bf16_t*)w; w += NE * 2;
  bf16_t* kbf = (bf16_t*)w; w += NE * 2;
  bf16_t* vbf = (bf16_t*)w; w += NE * 2;
  bf16_t* WqT = (bf16_t*)w; w += WE * 2;
  bf16_t* WkT = (bf16_t*)w; w += WE * 2;
  bf16_t* WvT = (bf16_t*)w; w += WE * 2;
  bf16_t* WoT = (bf16_t*)w; w += WE * 2;
  bf16_t* qhB = (bf16_t*)w; w += NE * 2;    // [B,H,S,D]
  bf16_t* khB = (bf16_t*)w; w += NE * 2;    // [B,H,S,D]
  bf16_t* vTb = (bf16_t*)w; w += NE * 2;    // [B,H,D,S]
  float*  tbl = (float*)w;  w += (long)B_ * H_ * S_ * 48 * 4;
  bf16_t* xcat = (bf16_t*)w; w += NE * 2;   // [B,S,HID]

  dim3 cg((unsigned)((NE + 255) / 256));
  k_cvt<<<cg, 256, 0, stream>>>(q, qbf, (int)NE);
  k_cvt<<<cg, 256, 0, stream>>>(k, kbf, (int)NE);
  k_cvt<<<cg, 256, 0, stream>>>(v, vbf, (int)NE);

  dim3 tg((unsigned)((WE + 255) / 256));
  k_transpose<<<tg, 256, 0, stream>>>(Wq, WqT, HID_, HID_);
  k_transpose<<<tg, 256, 0, stream>>>(Wk, WkT, HID_, HID_);
  k_transpose<<<tg, 256, 0, stream>>>(Wv, WvT, HID_, HID_);
  k_transpose<<<tg, 256, 0, stream>>>(Wo, WoT, HID_, HID_);

  dim3 gg((B_ * S_) / 64, HID_ / 64);       // (64, 8)
  k_gemm<<<gg, 128, 0, stream>>>(qbf, WqT, bq, qhB, B_ * S_, HID_, HID_, 0);
  k_gemm<<<gg, 128, 0, stream>>>(kbf, WkT, bk, khB, B_ * S_, HID_, HID_, 0);
  k_gemm<<<gg, 128, 0, stream>>>(vbf, WvT, bv, vTb, B_ * S_, HID_, HID_, 2);

  k_scores<<<dim3(B_ * H_ * S_), 64, 0, stream>>>(qhB, khB, qhopE, qedgeE,
                                                  khopE, kedgeE, tbl);

  size_t shmem = 64 * 512 * 4 + 64 * 48 * 4;   // 143,360 B dynamic LDS
  k_attn<<<dim3(B_ * H_ * (S_ / 64)), 256, shmem, stream>>>(
      qhB, khB, vTb, tbl, dist, edge, mask, focal, vhopE, vedgeE, xcat);

  k_gemm<<<gg, 128, 0, stream>>>(xcat, WoT, bo, d_out, B_ * S_, HID_, HID_, 1);
}
